// EgoRNN_31722628448657
// MI455X (gfx1250) — matrix operand order repaired and checked by reading, not compiled, not run on hardware
//
#include <hip/hip_runtime.h>

// ---------------- Problem dims (fixed by reference setup_inputs) ------------
constexpr int B_ = 64, T_ = 512, D_ = 128, N_ = 512, O_ = 256;
constexpr int ROWS_ = B_ * T_;          // 32768
constexpr float EPS_ = 1e-10f;

// ---------------- WMMA vector types ----------------------------------------
typedef __attribute__((ext_vector_type(8)))  float  v8f;
typedef __attribute__((ext_vector_type(2)))  float  v2f;
typedef __attribute__((ext_vector_type(16))) __bf16 v16bf;

union BF16x16 { v16bf v; unsigned int u[8]; uint4 q[2]; };

__device__ __forceinline__ unsigned int f2bf_rne(float f) {
  unsigned int u = __float_as_uint(f);
  u += 0x7FFFu + ((u >> 16) & 1u);     // round-to-nearest-even to bf16
  return u >> 16;
}
__device__ __forceinline__ unsigned int pack_bf16(float lo, float hi) {
  return f2bf_rne(lo) | (f2bf_rne(hi) << 16);
}

// ============================================================================
// P1: x (f32 row-major) -> xb (bf16 row-major), packed 2/u32
// ============================================================================
__global__ void k_xbf(const float* __restrict__ x, unsigned int* __restrict__ xb) {
  const int idx = blockIdx.x * blockDim.x + threadIdx.x;
  if (idx >= ROWS_ * D_ / 2) return;
  xb[idx] = pack_bf16(x[2 * idx], x[2 * idx + 1]);
}

// ============================================================================
// P2: generic bf16 B-panel pack for W[K x C]:
//   Bp[((kblk*tilesN + tn)*32 + lane)*8 + v] =
//       pack( W[(kblk*32 + 16g + 2v)*C + tn*16+mr], W[.. +1 ..] )
// so each wave's B tile (32x16 bf16) = 2 x b128 loads per lane.
// ============================================================================
__global__ void k_pack_b(const float* __restrict__ W, unsigned int* __restrict__ Bp,
                         int Kdim, int Cols) {
  const int idx = blockIdx.x * blockDim.x + threadIdx.x;
  const int tilesN = Cols / 16;
  if (idx >= (Kdim / 32) * tilesN * 256) return;
  const int v = idx & 7;
  const int lane = (idx >> 3) & 31;
  const int t2 = idx >> 8;
  const int tn = t2 % tilesN, kblk = t2 / tilesN;
  const int g = lane >> 4, mr = lane & 15;
  const int n = tn * 16 + mr;
  const int k = kblk * 32 + 16 * g + 2 * v;
  Bp[idx] = pack_bf16(W[k * Cols + n], W[(k + 1) * Cols + n]);
}

// ============================================================================
// P3: Wh (f32 [N x N]) -> v2f panels for V_WMMA_F32_16X16X4_F32 B operand:
//   Whp[(k4*32 + tn)*32 + lane] = { Wh[(4k4+2g)*N + n], Wh[(4k4+2g+1)*N + n] }
// ============================================================================
__global__ void k_pack_wh(const float* __restrict__ Wh, v2f* __restrict__ Whp) {
  const int idx = blockIdx.x * blockDim.x + threadIdx.x;
  if (idx >= (N_ / 4) * (N_ / 16) * 32) return;
  const int lane = idx & 31;
  const int t2 = idx >> 5;
  const int tn = t2 % (N_ / 16), k4 = t2 / (N_ / 16);
  const int g = lane >> 4, mr = lane & 15;
  const int n = tn * 16 + mr;
  const int k = 4 * k4 + 2 * g;
  v2f val; val[0] = Wh[k * N_ + n]; val[1] = Wh[(k + 1) * N_ + n];
  Whp[idx] = val;
}

// ============================================================================
// K1: states[row][n] = x[row][:] @ Wx + b  (bf16 WMMA, K=128)
// Inner loop: 2xb128 (A) + 2xb128 (B) + 1 wmma.
// ============================================================================
__global__ void k_xw(const uint4* __restrict__ Aq, const uint4* __restrict__ Bq,
                     const float* __restrict__ bias, float* __restrict__ states) {
  const int wid  = (blockIdx.x * blockDim.x + threadIdx.x) >> 5;
  const int lane = threadIdx.x & 31;
  const int g = lane >> 4, mr = lane & 15;
  const int tm = wid >> 5, tn = wid & 31;          // 2048 x 32 tiles
  if (tm >= ROWS_ / 16) return;
  const int n = tn * 16 + mr;
  const int arow = tm * 16 + mr;                   // A row = output row group

  v8f acc = {};
#pragma unroll
  for (int kblk = 0; kblk < D_ / 32; ++kblk) {
    BF16x16 a, b;
    const int ai = arow * (D_ / 8) + kblk * 4 + 2 * g; // uint4 index, 16B aligned
    a.q[0] = Aq[ai];
    a.q[1] = Aq[ai + 1];
    const int bi = ((kblk * 32 + tn) * 32 + lane) * 2;
    b.q[0] = Bq[bi];
    b.q[1] = Bq[bi + 1];
    acc = __builtin_amdgcn_wmma_f32_16x16x32_bf16(false, a.v, false, b.v,
                                                  (short)0, acc, false, false);
  }
  const float bn = bias[n];
#pragma unroll
  for (int v = 0; v < 8; ++v)                      // D: VGPR v -> row M=v+8g
    states[(tm * 16 + v + 8 * g) * N_ + n] = acc[v] + bn;
}

// ============================================================================
// K2: recurrence step t: states[b][t][:] = relu(xw_t + h_{t-1} @ Wh)
// f32 WMMA 16x16x4, K=512. One wave handles its N-column tile for ALL 4
// batch tiles: 4 independent accumulators hide WMMA latency on the serial
// chain, and each Wh panel b64 load feeds 4 WMMAs (Wh read once per step).
// Dual-stores h as f32 (recursion) and bf16 (feeds output GEMM).
// ============================================================================
__global__ void k_step(const v2f* __restrict__ Whp, const float* __restrict__ h0,
                       float* __restrict__ states, unsigned short* __restrict__ sbf,
                       int t) {
  const int wid  = (blockIdx.x * blockDim.x + threadIdx.x) >> 5;  // 0..31
  const int lane = threadIdx.x & 31;
  const int g = lane >> 4, mr = lane & 15;
  const int tn = wid;                      // N tile (0..31)
  const int n = tn * 16 + mr;

  // A-row pointers: batch rows brow = tm*16 + mr, reading h_{t-1}
  const float* ap[4];
#pragma unroll
  for (int tm = 0; tm < 4; ++tm) {
    const int brow = tm * 16 + mr;
    const float* hp = (t == 0) ? h0 : (states + (brow * T_ + (t - 1)) * N_);
    ap[tm] = hp + 2 * g;
  }
  const v2f* wp = Whp + tn * 32 + lane;

  // C preloaded with xw_t tiles (C/D layout: VGPR v -> row M = v + 8g)
  v8f acc[4];
#pragma unroll
  for (int tm = 0; tm < 4; ++tm)
#pragma unroll
    for (int v = 0; v < 8; ++v)
      acc[tm][v] = states[((tm * 16 + v + 8 * g) * T_ + t) * N_ + n];

#pragma unroll 4
  for (int k4 = 0; k4 < N_ / 4; ++k4) {
    const v2f b = *wp;  wp += (N_ / 16) * 32;
    v2f a0 = *(const v2f*)ap[0];  ap[0] += 4;
    v2f a1 = *(const v2f*)ap[1];  ap[1] += 4;
    v2f a2 = *(const v2f*)ap[2];  ap[2] += 4;
    v2f a3 = *(const v2f*)ap[3];  ap[3] += 4;
    acc[0] = __builtin_amdgcn_wmma_f32_16x16x4_f32(false, a0, false, b,
                                                   (short)0, acc[0], false, false);
    acc[1] = __builtin_amdgcn_wmma_f32_16x16x4_f32(false, a1, false, b,
                                                   (short)0, acc[1], false, false);
    acc[2] = __builtin_amdgcn_wmma_f32_16x16x4_f32(false, a2, false, b,
                                                   (short)0, acc[2], false, false);
    acc[3] = __builtin_amdgcn_wmma_f32_16x16x4_f32(false, a3, false, b,
                                                   (short)0, acc[3], false, false);
  }
#pragma unroll
  for (int tm = 0; tm < 4; ++tm)
#pragma unroll
    for (int v = 0; v < 8; ++v) {
      const int row = tm * 16 + v + 8 * g;
      const float h = fmaxf(acc[tm][v], 0.0f);
      states[(row * T_ + t) * N_ + n] = h;
      sbf[(row * T_ + t) * N_ + n] = (unsigned short)f2bf_rne(h);
    }
}

// ============================================================================
// K3: outs[row][o] = relu( states_bf16[row][:] @ Wo + bo )  (bf16 WMMA, K=512)
// ============================================================================
__global__ void k_out(const uint4* __restrict__ Aq, const uint4* __restrict__ Bq,
                      const float* __restrict__ bo, float* __restrict__ outs) {
  const int wid  = (blockIdx.x * blockDim.x + threadIdx.x) >> 5;
  const int lane = threadIdx.x & 31;
  const int g = lane >> 4, mr = lane & 15;
  const int tm = wid >> 4, tn = wid & 15;          // 2048 x 16 tiles
  if (tm >= ROWS_ / 16) return;
  const int n = tn * 16 + mr;
  const int arow = tm * 16 + mr;

  v8f acc = {};
#pragma unroll 4
  for (int kblk = 0; kblk < N_ / 32; ++kblk) {
    BF16x16 a, b;
    const int ai = arow * (N_ / 8) + kblk * 4 + 2 * g;
    a.q[0] = Aq[ai];
    a.q[1] = Aq[ai + 1];
    const int bi = ((kblk * 16 + tn) * 32 + lane) * 2;
    b.q[0] = Bq[bi];
    b.q[1] = Bq[bi + 1];
    acc = __builtin_amdgcn_wmma_f32_16x16x32_bf16(false, a.v, false, b.v,
                                                  (short)0, acc, false, false);
  }
  const float bn = bo[n];
#pragma unroll
  for (int v = 0; v < 8; ++v)
    outs[(tm * 16 + v + 8 * g) * O_ + n] = fmaxf(acc[v] + bn, 0.0f);
}

// ============================================================================
// K4: denomT[b][o] = sum_t outs[b][t][o] + T*EPS   (thread per (b,o))
// ============================================================================
__global__ void k_denomT(const float* __restrict__ outs, float* __restrict__ denomT) {
  const int idx = blockIdx.x * blockDim.x + threadIdx.x;
  if (idx >= B_ * O_) return;
  const int b = idx / O_, o = idx % O_;
  const float* __restrict__ p = outs + b * T_ * O_ + o;
  float s = 0.0f;
  for (int t = 0; t < T_; ++t) s += p[t * O_];
  denomT[idx] = s + (float)T_ * EPS_;
}

// ============================================================================
// K5: denomO[row] = sum_o outs[row][o] + O*EPS   (wave per row, shfl reduce)
// ============================================================================
__global__ void k_denomO(const float* __restrict__ outs, float* __restrict__ denomO) {
  const int wid  = (blockIdx.x * blockDim.x + threadIdx.x) >> 5;
  const int lane = threadIdx.x & 31;
  if (wid >= ROWS_) return;
  const float* __restrict__ p = outs + wid * O_;
  float s = 0.0f;
  for (int o = lane; o < O_; o += 32) s += p[o];
  for (int off = 16; off > 0; off >>= 1) s += __shfl_xor(s, off, 32);
  if (lane == 0) denomO[wid] = s + (float)O_ * EPS_;
}

// ============================================================================
// K6: centers[b][o][d] = (sum_t r[b][t][d]*outs[b][t][o]) / denomT[b][o]
// ============================================================================
__global__ void k_centers(const float* __restrict__ outs, const float* __restrict__ r,
                          const float* __restrict__ denomT, float* __restrict__ centers) {
  const int idx = blockIdx.x * blockDim.x + threadIdx.x;
  if (idx >= B_ * O_) return;
  const int b = idx / O_, o = idx % O_;
  const float* __restrict__ p  = outs + b * T_ * O_ + o;
  const float* __restrict__ rr = r + b * T_ * 2;
  float a0 = 0.0f, a1 = 0.0f;
  for (int t = 0; t < T_; ++t) {
    const float w = p[t * O_];
    a0 += rr[2 * t] * w;
    a1 += rr[2 * t + 1] * w;
  }
  const float inv = 1.0f / denomT[idx];
  centers[2 * idx]     = a0 * inv;
  centers[2 * idx + 1] = a1 * inv;
}

// ============================================================================
// K7: out[b][t][d] = (sum_o outs[row][o]*centers[b][o][d]) / denomO[row]
// ============================================================================
__global__ void k_pos(const float* __restrict__ outs, const float* __restrict__ centers,
                      const float* __restrict__ denomO, float* __restrict__ out) {
  const int wid  = (blockIdx.x * blockDim.x + threadIdx.x) >> 5;
  const int lane = threadIdx.x & 31;
  if (wid >= ROWS_) return;
  const int b = wid / T_;
  const float* __restrict__ p = outs + wid * O_;
  const float* __restrict__ c = centers + b * O_ * 2;
  float a0 = 0.0f, a1 = 0.0f;
  for (int o = lane; o < O_; o += 32) {
    const float w = p[o];
    a0 += w * c[2 * o];
    a1 += w * c[2 * o + 1];
  }
  for (int off = 16; off > 0; off >>= 1) {
    a0 += __shfl_xor(a0, off, 32);
    a1 += __shfl_xor(a1, off, 32);
  }
  if (lane == 0) {
    const float inv = 1.0f / denomO[wid];
    out[2 * wid]     = a0 * inv;
    out[2 * wid + 1] = a1 * inv;
  }
}

// ============================================================================
extern "C" void kernel_launch(void* const* d_in, const int* in_sizes, int n_in,
                              void* d_out, int out_size, void* d_ws, size_t ws_size,
                              hipStream_t stream) {
  const float* x    = (const float*)d_in[0];  // [B,T,D]
  const float* r    = (const float*)d_in[1];  // [B,T,2]
  const float* h0   = (const float*)d_in[2];  // [1,N]
  const float* Wx   = (const float*)d_in[3];  // [D,N]
  const float* Wh   = (const float*)d_in[4];  // [N,N]
  const float* bias = (const float*)d_in[5];  // [N]
  const float* Wo   = (const float*)d_in[6];  // [N,O]
  const float* bo   = (const float*)d_in[7];  // [O]
  float* out = (float*)d_out;

  // ---- workspace layout (bytes) -------------------------------------------
  char* w = (char*)d_ws;
  float*          states = (float*)w;                    w += (size_t)ROWS_ * N_ * 4;   // 64 MB
  unsigned short* sbf    = (unsigned short*)w;           w += (size_t)ROWS_ * N_ * 2;   // 32 MB
  float*          outs   = (float*)w;                    w += (size_t)ROWS_ * O_ * 4;   // 32 MB
  // xb aliases the outs region: dead before k_out writes outs
  unsigned int*   xb     = (unsigned int*)outs;                                        // 8 MB
  unsigned int*   Wxp    = (unsigned int*)w;             w += (size_t)(D_/32)*(N_/16)*256 * 4;
  unsigned int*   Wop    = (unsigned int*)w;             w += (size_t)(N_/32)*(O_/16)*256 * 4;
  v2f*            Whp    = (v2f*)w;                      w += (size_t)(N_/4)*(N_/16)*32 * 8;
  float*          denomT = (float*)w;                    w += (size_t)B_ * O_ * 4;
  float*          denomO = (float*)w;                    w += (size_t)ROWS_ * 4;
  float*          centers= (float*)w;                    w += (size_t)B_ * O_ * 2 * 4;

  // ---- prepack (bandwidth-trivial one-shots) ------------------------------
  k_xbf    <<<ROWS_ * D_ / 2 / 256, 256, 0, stream>>>(x, xb);
  k_pack_b <<<(D_/32)*(N_/16), 256, 0, stream>>>(Wx, Wxp, D_, N_);
  k_pack_b <<<(N_/32)*(O_/16), 256, 0, stream>>>(Wo, Wop, N_, O_);
  k_pack_wh<<<(N_/4)*(N_/16)*32 / 256, 256, 0, stream>>>(Wh, Whp);

  // ---- 1) xw precompute: 2048x32 tiles, 8 waves/block ---------------------
  k_xw<<<(ROWS_/16)*(N_/16)/8, 256, 0, stream>>>((const uint4*)xb, (const uint4*)Wxp,
                                                 bias, states);

  // ---- 2) serial recurrence: 32 waves/step (4 acc each), launch = sync ----
  for (int t = 0; t < T_; ++t)
    k_step<<<4, 256, 0, stream>>>(Whp, h0, states, sbf, t);

  // ---- 3) output GEMM + relu ----------------------------------------------
  k_out<<<(ROWS_/16)*(O_/16)/8, 256, 0, stream>>>((const uint4*)sbf, (const uint4*)Wop,
                                                  bo, outs);

  // ---- 4-7) normalizations and tiny einsums -------------------------------
  k_denomT <<<(B_ * O_ + 255) / 256, 256, 0, stream>>>(outs, denomT);
  k_denomO <<<ROWS_ / 8,            256, 0, stream>>>(outs, denomO);
  k_centers<<<(B_ * O_ + 255) / 256, 256, 0, stream>>>(outs, r, denomT, centers);
  k_pos    <<<ROWS_ / 8,            256, 0, stream>>>(outs, centers, denomO, out);
}
